// BlocksparseDilatedAttention_90967407329532
// MI455X (gfx1250) — matrix operand order repaired and checked by reading, not compiled, not run on hardware
//
#include <hip/hip_runtime.h>
#include <hip/hip_bf16.h>

// ---------------------------------------------------------------------------
// BlocksparseDilatedAttention for MI455X (gfx1250, wave32, WMMA f16->f32)
//   B=2, S=8192, D=768, H=12 (hd=64), R=4, SEG=512, L=2048, n_seg=4
// Pipeline:
//   [zero out] + [gather/cvt x->f16] + [cvt W->f16]
//   -> [qkv proj GEMM f16]  -> [flash attn per seg/head] -> [out proj + scatter]
// All matrix math on v_wmma_f32_16x16x32_f16; all operands pre-converted f16
// so GEMM inner loops are pure b128-load + wmma (no f32->f16 cvt in hot loop).
// ---------------------------------------------------------------------------

typedef __attribute__((ext_vector_type(16))) _Float16 v16h;
typedef __attribute__((ext_vector_type(8)))  float    v8f;

#define BATCH 2
#define SEQ   8192
#define DM    768
#define H     12
#define HD    64
#define R     4
#define SEG   512
#define LTOK  (SEQ / R)          // 2048
#define NSEG  (LTOK / SEG)       // 4
#define D3    (3 * DM)           // 2304
#define RD    (R * DM)           // 3072
#define RB    (R * BATCH)        // 8
#define SCALE 0.125f             // 1/sqrt(64), exact in f16

static_assert(LTOK % 128 == 0 && D3 % 64 == 0 && DM % 64 == 0, "tiling");

// ---------------------------------------------------------------------------
// Kernel 0a: zero-fill output (B, S, R*D) — 3/4 stays zero structurally.
// ---------------------------------------------------------------------------
__global__ __launch_bounds__(256) void zero_out_kernel(float4* __restrict__ out,
                                                       long long n4) {
  long long i = (long long)blockIdx.x * blockDim.x + threadIdx.x;
  if (i < n4) {
    float4 z; z.x = 0.f; z.y = 0.f; z.z = 0.f; z.w = 0.f;
    out[i] = z;
  }
}

// ---------------------------------------------------------------------------
// Kernel 0b: dilated gather + f32->f16: xh[rb][l][d] = x[b][l*R+o][d]
// 8 elements per thread (two float4 reads, one uint4 write).
// ---------------------------------------------------------------------------
__global__ __launch_bounds__(256) void gather_cvt_x_kernel(
    const float* __restrict__ x, _Float16* __restrict__ xh) {
  const size_t idx = (size_t)blockIdx.x * blockDim.x + threadIdx.x;
  const size_t e8 = idx * 8;
  if (e8 >= (size_t)RB * LTOK * DM) return;
  const int d  = (int)(e8 % DM);
  const int l  = (int)((e8 / DM) % LTOK);
  const int rb = (int)(e8 / ((size_t)DM * LTOK));
  const int o = rb / BATCH, b = rb % BATCH;
  const float* __restrict__ src =
      x + ((size_t)b * SEQ + (size_t)l * R + o) * DM + d;
  const float4 f0 = reinterpret_cast<const float4*>(src)[0];
  const float4 f1 = reinterpret_cast<const float4*>(src)[1];
  union { uint4 u; _Float16 h[8]; } t;
  t.h[0] = (_Float16)f0.x; t.h[1] = (_Float16)f0.y;
  t.h[2] = (_Float16)f0.z; t.h[3] = (_Float16)f0.w;
  t.h[4] = (_Float16)f1.x; t.h[5] = (_Float16)f1.y;
  t.h[6] = (_Float16)f1.z; t.h[7] = (_Float16)f1.w;
  reinterpret_cast<uint4*>(xh + e8)[0] = t.u;
}

// ---------------------------------------------------------------------------
// Kernel 0c: generic f32->f16 convert (weights), 8 elems/thread.
// ---------------------------------------------------------------------------
__global__ __launch_bounds__(256) void cvt_f16_kernel(
    const float* __restrict__ src, _Float16* __restrict__ dst, long long n) {
  const size_t e8 = ((size_t)blockIdx.x * blockDim.x + threadIdx.x) * 8;
  if ((long long)e8 >= n) return;
  const float4 f0 = reinterpret_cast<const float4*>(src + e8)[0];
  const float4 f1 = reinterpret_cast<const float4*>(src + e8)[1];
  union { uint4 u; _Float16 h[8]; } t;
  t.h[0] = (_Float16)f0.x; t.h[1] = (_Float16)f0.y;
  t.h[2] = (_Float16)f0.z; t.h[3] = (_Float16)f0.w;
  t.h[4] = (_Float16)f1.x; t.h[5] = (_Float16)f1.y;
  t.h[6] = (_Float16)f1.z; t.h[7] = (_Float16)f1.w;
  reinterpret_cast<uint4*>(dst + e8)[0] = t.u;
}

// ---------------------------------------------------------------------------
// WMMA fragment loaders (row-major f16 source).
// A 16x32 tile: lanes 0-15 = row m, K {0..7,16..23}; lanes 16-31 K {8..15,24..31}
// B 32x16 tile: lane ll = col, K {0..7,16..23} for lanes 0-15; {8..15,24..31} for 16-31
// ---------------------------------------------------------------------------
__device__ __forceinline__ v16h load_a_f16(const _Float16* __restrict__ row,
                                           int k0, int hi) {
  v16h a;
#pragma unroll
  for (int j = 0; j < 8; ++j) {
    a[j]     = row[k0 + 8 * hi + j];
    a[j + 8] = row[k0 + 16 + 8 * hi + j];
  }
  return a;
}
__device__ __forceinline__ v16h load_bt_f16(const _Float16* __restrict__ row,
                                            int k0, int hi) {
  v16h b;
#pragma unroll
  for (int j = 0; j < 8; ++j) {
    b[j]     = row[k0 + 8 * hi + j];
    b[j + 8] = row[k0 + 16 + 8 * hi + j];
  }
  return b;
}

// ---------------------------------------------------------------------------
// Kernel 1: QKV projection. QKV[rb][l][e] = xh[rb][l][:] . Wh[o][e][:] + bqkv
// Grid: (RB, L/128, 3D/64), block 128 (4 waves); wave computes 32x64.
// ---------------------------------------------------------------------------
__global__ __launch_bounds__(128) void qkv_proj_kernel(
    const _Float16* __restrict__ xh, const _Float16* __restrict__ Wh,
    const float* __restrict__ bqkv, _Float16* __restrict__ qkv) {
  const int rb = blockIdx.x;
  const int o = rb / BATCH;
  const int wave = threadIdx.x >> 5;
  const int lane = threadIdx.x & 31;
  const int ll = lane & 15, hi = lane >> 4;
  const int m0 = blockIdx.y * 128 + wave * 32;
  const int n0 = blockIdx.z * 64;

  const _Float16* __restrict__ W = Wh + (size_t)o * D3 * DM;
  const _Float16* __restrict__ ar0 = xh + ((size_t)rb * LTOK + m0 + ll) * DM;
  const _Float16* __restrict__ ar1 = ar0 + (size_t)16 * DM;

  v8f acc0[4] = {}, acc1[4] = {};
  for (int k0 = 0; k0 < DM; k0 += 32) {
    const v16h a0 = load_a_f16(ar0, k0, hi);
    const v16h a1 = load_a_f16(ar1, k0, hi);
#pragma unroll
    for (int t = 0; t < 4; ++t) {
      const v16h bf =
          load_bt_f16(W + (size_t)(n0 + t * 16 + ll) * DM, k0, hi);
      acc0[t] = __builtin_amdgcn_wmma_f32_16x16x32_f16(
          false, a0, false, bf, (short)0, acc0[t], false, false);
      acc1[t] = __builtin_amdgcn_wmma_f32_16x16x32_f16(
          false, a1, false, bf, (short)0, acc1[t], false, false);
    }
  }
#pragma unroll
  for (int t = 0; t < 4; ++t) {
    const int n = n0 + t * 16 + ll;
    const float bias = bqkv[o * D3 + n];
#pragma unroll
    for (int i = 0; i < 8; ++i) {
      const int m = m0 + i + 8 * hi;            // C/D: row = i + 8*hi, col = ll
      qkv[((size_t)rb * LTOK + m) * D3 + n]      = (_Float16)(acc0[t][i] + bias);
      qkv[((size_t)rb * LTOK + m + 16) * D3 + n] = (_Float16)(acc1[t][i] + bias);
    }
  }
}

// ---------------------------------------------------------------------------
// Kernel 2: flash attention per (rb, seg, head, q-block of 64).
//   Block = 4 waves; wave owns 16 Q rows. K/V streamed in 32-key LDS chunks;
//   V staged transposed so every LDS B-fragment read is two 16B contiguous runs.
// Grid: (RB, NSEG*H, SEG/64), block = 128.
// ---------------------------------------------------------------------------
__global__ __launch_bounds__(128) void attn_kernel(
    const _Float16* __restrict__ qkv, _Float16* __restrict__ ctx) {
  __shared__ __align__(16) _Float16 Klds[32][HD];        // [key][d]   4 KB
  __shared__ __align__(16) _Float16 Vt[HD][32];          // [d][key]   4 KB
  __shared__ __align__(16) _Float16 Plds[4][16][32];     // per-wave   4 KB

  const int rb  = blockIdx.x;
  const int seg = blockIdx.y / H;
  const int h   = blockIdx.y % H;
  const int wave = threadIdx.x >> 5;
  const int lane = threadIdx.x & 31;
  const int ll = lane & 15, hi = lane >> 4;
  const int q0 = blockIdx.z * 64 + wave * 16;

  const size_t seg_row = (size_t)rb * LTOK + (size_t)seg * SEG;

  // Resident Q A-fragments with softmax scale folded in (0.125 exact in f16)
  v16h qa[2];
  {
    const _Float16* __restrict__ qrow =
        qkv + (seg_row + q0 + ll) * D3 + h * HD;
    const _Float16 sc = (_Float16)SCALE;
#pragma unroll
    for (int f = 0; f < 2; ++f)
#pragma unroll
      for (int j = 0; j < 8; ++j) {
        qa[f][j]     = qrow[32 * f + 8 * hi + j] * sc;
        qa[f][j + 8] = qrow[32 * f + 16 + 8 * hi + j] * sc;
      }
  }

  v8f   oacc[4] = {};
  float mrun[8], lrun[8];
#pragma unroll
  for (int i = 0; i < 8; ++i) { mrun[i] = -3.0e30f; lrun[i] = 0.f; }

  for (int key0 = 0; key0 < SEG; key0 += 32) {
    __syncthreads();  // previous chunk fully consumed
    // Cooperative stage: K row-major, V transposed. 512 uint4 reads total.
    for (int i = threadIdx.x; i < 512; i += 128) {
      const int mat = i >> 8;                   // 0 = K, 1 = V
      const int rem = i & 255;
      const int row = rem >> 3;                 // key within chunk
      const int c4  = rem & 7;                  // uint4 (8 halves) within row
      const _Float16* src =
          qkv + (seg_row + key0 + row) * D3 + (mat ? 2 * DM : DM) + h * HD;
      union { uint4 u; _Float16 hh[8]; } t;
      t.u = reinterpret_cast<const uint4*>(src)[c4];
      if (mat == 0) {
        reinterpret_cast<uint4*>(&Klds[row][0])[c4] = t.u;
      } else {
        const int d0 = c4 * 8;
#pragma unroll
        for (int jj = 0; jj < 8; ++jj) Vt[d0 + jj][row] = t.hh[jj];
      }
    }
    __syncthreads();

    // Scores: two 16x16 tiles (keys key0..+15, +16..+31), K-dim = 64
    v8f s0 = {}, s1 = {};
#pragma unroll
    for (int f = 0; f < 2; ++f) {
      v16h kb = load_bt_f16(&Klds[ll][0], 32 * f, hi);
      s0 = __builtin_amdgcn_wmma_f32_16x16x32_f16(false, qa[f], false, kb,
                                                  (short)0, s0, false, false);
      kb = load_bt_f16(&Klds[16 + ll][0], 32 * f, hi);
      s1 = __builtin_amdgcn_wmma_f32_16x16x32_f16(false, qa[f], false, kb,
                                                  (short)0, s1, false, false);
    }

    // Online softmax. Lane ll holds column ll of rows {i+8*hi}; a 16-lane
    // xor-tree gives per-row stats (halves own disjoint row groups).
    float mb[8];
#pragma unroll
    for (int i = 0; i < 8; ++i) mb[i] = fmaxf(s0[i], s1[i]);
#pragma unroll
    for (int off = 8; off >= 1; off >>= 1)
#pragma unroll
      for (int i = 0; i < 8; ++i)
        mb[i] = fmaxf(mb[i], __shfl_xor(mb[i], off, 32));

    float alpha[8];
#pragma unroll
    for (int i = 0; i < 8; ++i) {
      const float mn = fmaxf(mrun[i], mb[i]);
      alpha[i] = __expf(mrun[i] - mn);
      mrun[i] = mn;
    }
#pragma unroll
    for (int i = 0; i < 8; ++i) {
      s0[i] = __expf(s0[i] - mrun[i]);
      s1[i] = __expf(s1[i] - mrun[i]);
    }
    float rs[8];
#pragma unroll
    for (int i = 0; i < 8; ++i) rs[i] = s0[i] + s1[i];
#pragma unroll
    for (int off = 8; off >= 1; off >>= 1)
#pragma unroll
      for (int i = 0; i < 8; ++i) rs[i] += __shfl_xor(rs[i], off, 32);
#pragma unroll
    for (int i = 0; i < 8; ++i) lrun[i] = lrun[i] * alpha[i] + rs[i];
#pragma unroll
    for (int t = 0; t < 4; ++t)
#pragma unroll
      for (int i = 0; i < 8; ++i) oacc[t][i] *= alpha[i];

    // Re-layout P (D-tile) -> A-fragment via per-wave LDS slab
#pragma unroll
    for (int i = 0; i < 8; ++i) {
      Plds[wave][i + 8 * hi][ll]      = (_Float16)s0[i];
      Plds[wave][i + 8 * hi][16 + ll] = (_Float16)s1[i];
    }
    const v16h pa = load_a_f16(&Plds[wave][ll][0], 0, hi);

    // ctx += P(16x32) x V(32x64): B[k][n] = Vt[n][k] (contiguous per lane)
#pragma unroll
    for (int t = 0; t < 4; ++t) {
      const v16h vb = load_bt_f16(&Vt[t * 16 + ll][0], 0, hi);
      oacc[t] = __builtin_amdgcn_wmma_f32_16x16x32_f16(
          false, pa, false, vb, (short)0, oacc[t], false, false);
    }
  }

  // Normalize and store ctx as f16: [rb][L][D]
  float inv[8];
#pragma unroll
  for (int i = 0; i < 8; ++i) inv[i] = 1.0f / lrun[i];
#pragma unroll
  for (int t = 0; t < 4; ++t)
#pragma unroll
    for (int i = 0; i < 8; ++i) {
      const int m = q0 + i + 8 * hi;
      ctx[(seg_row + m) * DM + h * HD + t * 16 + ll] =
          (_Float16)(oacc[t][i] * inv[i]);
    }
}

// ---------------------------------------------------------------------------
// Kernel 3: output projection + dilated scatter.
//   out[b][l*R+o][o*D + n] = ctx[rb][l][:] . Wouth[o][n][:] + bout[o][n]
// Grid: (RB, L/128, D/64), block 128; wave computes 32x64.
// ---------------------------------------------------------------------------
__global__ __launch_bounds__(128) void out_proj_kernel(
    const _Float16* __restrict__ ctx, const _Float16* __restrict__ Wouth,
    const float* __restrict__ bout, float* __restrict__ out) {
  const int rb = blockIdx.x;
  const int o = rb / BATCH, b = rb % BATCH;
  const int wave = threadIdx.x >> 5;
  const int lane = threadIdx.x & 31;
  const int ll = lane & 15, hi = lane >> 4;
  const int m0 = blockIdx.y * 128 + wave * 32;
  const int n0 = blockIdx.z * 64;

  const _Float16* __restrict__ W = Wouth + (size_t)o * DM * DM;
  const _Float16* __restrict__ ar0 = ctx + ((size_t)rb * LTOK + m0 + ll) * DM;
  const _Float16* __restrict__ ar1 = ar0 + (size_t)16 * DM;

  v8f acc0[4] = {}, acc1[4] = {};
  for (int k0 = 0; k0 < DM; k0 += 32) {
    const v16h a0 = load_a_f16(ar0, k0, hi);
    const v16h a1 = load_a_f16(ar1, k0, hi);
#pragma unroll
    for (int t = 0; t < 4; ++t) {
      const v16h bf =
          load_bt_f16(W + (size_t)(n0 + t * 16 + ll) * DM, k0, hi);
      acc0[t] = __builtin_amdgcn_wmma_f32_16x16x32_f16(
          false, a0, false, bf, (short)0, acc0[t], false, false);
      acc1[t] = __builtin_amdgcn_wmma_f32_16x16x32_f16(
          false, a1, false, bf, (short)0, acc1[t], false, false);
    }
  }
#pragma unroll
  for (int t = 0; t < 4; ++t) {
    const int n = n0 + t * 16 + ll;
    const float bias = bout[o * DM + n];
#pragma unroll
    for (int i = 0; i < 8; ++i) {
      const int l0 = m0 + i + 8 * hi;
#pragma unroll
      for (int u = 0; u < 2; ++u) {
        const int l = l0 + 16 * u;
        const long long s = (long long)l * R + o;          // dilated scatter
        out[((size_t)b * SEQ + s) * RD + o * DM + n] =
            (u ? acc1[t][i] : acc0[t][i]) + bias;
      }
    }
  }
}

// ---------------------------------------------------------------------------
// Host launcher
// ---------------------------------------------------------------------------
extern "C" void kernel_launch(void* const* d_in, const int* in_sizes, int n_in,
                              void* d_out, int out_size, void* d_ws,
                              size_t ws_size, hipStream_t stream) {
  const float* x    = (const float*)d_in[0];
  const float* Wqkv = (const float*)d_in[1];
  const float* bqkv = (const float*)d_in[2];
  const float* Wout = (const float*)d_in[3];
  const float* bout = (const float*)d_in[4];
  float* out = (float*)d_out;

  // Workspace layout (f16):
  //   [qkv 75.5MB][xh/ctx 25.2MB (aliased: xh dead after k1)][Wqkvh][Wouth]
  const size_t qkv_elems = (size_t)RB * LTOK * D3;             // 37.7M
  const size_t xh_elems  = (size_t)RB * LTOK * DM;             // 12.6M
  const size_t wq_elems  = (size_t)R * D3 * DM;                // 7.08M
  const size_t wo_elems  = (size_t)R * DM * DM;                // 2.36M
  _Float16* qkv   = (_Float16*)d_ws;
  _Float16* xh    = qkv + qkv_elems;
  _Float16* ctx   = xh;                                        // alias (safe)
  _Float16* Wqkvh = xh + xh_elems;
  _Float16* Wouth = Wqkvh + wq_elems;

  // 0) zero output + precision prep (independent, all on stream)
  const long long n4 = (long long)BATCH * SEQ * RD / 4;
  zero_out_kernel<<<(unsigned)((n4 + 255) / 256), 256, 0, stream>>>(
      (float4*)out, n4);
  gather_cvt_x_kernel<<<(unsigned)(xh_elems / 8 / 256 + 1), 256, 0, stream>>>(
      x, xh);
  cvt_f16_kernel<<<(unsigned)(wq_elems / 8 / 256 + 1), 256, 0, stream>>>(
      Wqkv, Wqkvh, (long long)wq_elems);
  cvt_f16_kernel<<<(unsigned)(wo_elems / 8 / 256 + 1), 256, 0, stream>>>(
      Wout, Wouth, (long long)wo_elems);

  // 1) QKV projection
  qkv_proj_kernel<<<dim3(RB, LTOK / 128, D3 / 64), 128, 0, stream>>>(
      xh, Wqkvh, bqkv, qkv);

  // 2) segmented flash attention
  attn_kernel<<<dim3(RB, NSEG * H, SEG / 64), 128, 0, stream>>>(qkv, ctx);

  // 3) output projection + dilated scatter
  out_proj_kernel<<<dim3(RB, LTOK / 128, DM / 64), 128, 0, stream>>>(
      ctx, Wouth, bout, out);
}
